// MPSGNN_71313636982970
// MI455X (gfx1250) — matrix operand order repaired and checked by reading, not compiled, not run on hardware
//
#include <hip/hip_runtime.h>
#include <hip/hip_bf16.h>

#define DD 128

typedef __attribute__((ext_vector_type(16))) __bf16          v16bf_t;
typedef __attribute__((ext_vector_type(8)))  float           v8f_t;
typedef __attribute__((ext_vector_type(8)))  unsigned short  u16x8_t;

__device__ __forceinline__ unsigned short f2bf(float f) {
    unsigned u = __float_as_uint(f);
    u += 0x7FFFu + ((u >> 16) & 1u);   // round-to-nearest-even
    return (unsigned short)(u >> 16);
}

// ---------------------------------------------------------------------------
// Pack one f32 [K=128][N=128] weight matrix into bf16 in WMMA B-fragment order:
//   out[(((tn*4 + kx)*2 + half)*16 + r)*16 + e] = bf16(W[kx*32+half*16+e][tn*16+r])
// so each lane's 16-element fragment is two contiguous b128 loads.
// ---------------------------------------------------------------------------
__global__ void pack_w_kernel(const float* __restrict__ in,
                              unsigned short* __restrict__ out) {
    int i = blockIdx.x * blockDim.x + threadIdx.x;   // 0 .. 16383
    int e    = i & 15;
    int r    = (i >> 4) & 15;
    int half = (i >> 8) & 1;
    int kx   = (i >> 9) & 3;
    int tn   = i >> 11;
    int k = kx * 32 + half * 16 + e;
    int c = tn * 16 + r;
    out[i] = f2bf(in[k * DD + c]);
}

__global__ void zero_kernel(float* __restrict__ p, long n) {
    long i = (long)blockIdx.x * blockDim.x + threadIdx.x;
    if (i < n) p[i] = 0.0f;
}

// ---------------------------------------------------------------------------
// Edge scatter: agg[dst] += x[src] (32 lanes/edge, float4 per lane), cnt[dst]++
// ---------------------------------------------------------------------------
__global__ void scatter_accum_kernel(const float* __restrict__ x,
                                     const int* __restrict__ src,
                                     const int* __restrict__ dst,
                                     float* __restrict__ agg,
                                     float* __restrict__ cnt, int nedges) {
    long gid  = (long)blockIdx.x * blockDim.x + threadIdx.x;
    long e    = gid >> 5;
    int  lane = (int)(gid & 31);
    if (e >= nedges) return;
    int s = src[e], d = dst[e];
    const float4 xv = *(const float4*)(x + (long)s * DD + lane * 4);
    float* ap = agg + (long)d * DD + lane * 4;
    atomicAdd(ap + 0, xv.x);
    atomicAdd(ap + 1, xv.y);
    atomicAdd(ap + 2, xv.z);
    atomicAdd(ap + 3, xv.w);
    if (lane == 0) atomicAdd(cnt + d, 1.0f);
}

// ---------------------------------------------------------------------------
// Fused WMMA GEMM: out[r,:] = act( scale(r)*A1[r,:] @ W1 + (A2? A2[r,:]@W2 :0) + b )
//  - compile-time specialized on HASA2 / HASCNT / RELU: straight-line k-loop,
//    no scalar branches between WMMAs, scheduler free to pipeline loads
//  - one block (256 thr / 8 waves) per 16-row m-tile; wave w computes n-tile w
//  - A tiles staged once in LDS as bf16 (8x reuse across waves), rows padded
//    by 8 shorts -> bank stride 68 % 64 = 4: conflict-free b128 ds reads
//  - weights pre-packed in fragment order -> two global b128 loads per frag
//  - row indices clamped (not predicated): WMMA executes with full EXEC;
//    only the epilogue stores are guarded
// ---------------------------------------------------------------------------
#define APITCH 136   // 128 + 8 pad (shorts)

template <bool HASA2, bool HASCNT, bool RELU>
__global__ void gemm_bias_act_kernel(const float* __restrict__ A1,
                                     const float* __restrict__ cnt,
                                     const float* __restrict__ A2,
                                     const unsigned short* __restrict__ W1p,
                                     const unsigned short* __restrict__ W2p,
                                     const float* __restrict__ bias,
                                     float* __restrict__ out,
                                     int nrows) {
    __shared__ __align__(16) unsigned short a1s[16 * APITCH];
    __shared__ __align__(16) unsigned short a2s[HASA2 ? 16 * APITCH : 8];
    const int tid = threadIdx.x;
    const int m0  = blockIdx.x << 4;

    // ---- stage A tile(s) into LDS as bf16 (each thread: 8 elements) ----
    {
        const int srow = tid >> 4;
        const int scol = (tid & 15) << 3;
        int grow = m0 + srow;
        if (grow >= nrows) grow = nrows - 1;          // clamp, no divergence
        float sc = 1.0f;
        if (HASCNT) sc = 1.0f / fmaxf(cnt[grow], 1.0f);  // fused mean divide
        const float4 xa = *(const float4*)(A1 + (long)grow * DD + scol);
        const float4 xb = *(const float4*)(A1 + (long)grow * DD + scol + 4);
        union { u16x8_t v; unsigned short u[8]; } p;
        p.u[0] = f2bf(xa.x * sc); p.u[1] = f2bf(xa.y * sc);
        p.u[2] = f2bf(xa.z * sc); p.u[3] = f2bf(xa.w * sc);
        p.u[4] = f2bf(xb.x * sc); p.u[5] = f2bf(xb.y * sc);
        p.u[6] = f2bf(xb.z * sc); p.u[7] = f2bf(xb.w * sc);
        *(u16x8_t*)(a1s + srow * APITCH + scol) = p.v;
        if (HASA2) {
            const float4 ya = *(const float4*)(A2 + (long)grow * DD + scol);
            const float4 yb = *(const float4*)(A2 + (long)grow * DD + scol + 4);
            p.u[0] = f2bf(ya.x); p.u[1] = f2bf(ya.y);
            p.u[2] = f2bf(ya.z); p.u[3] = f2bf(ya.w);
            p.u[4] = f2bf(yb.x); p.u[5] = f2bf(yb.y);
            p.u[6] = f2bf(yb.z); p.u[7] = f2bf(yb.w);
            *(u16x8_t*)(a2s + srow * APITCH + scol) = p.v;
        }
    }
    __syncthreads();

    // ---- per-wave 16x16 tile: 4 K-steps of 32, bf16 WMMA, f32 accumulate ----
    const int wave = tid >> 5;
    const int lane = tid & 31;
    const int half = lane >> 4;
    const int r    = lane & 15;
    const int n0   = wave << 4;

    v8f_t acc = {};
#pragma unroll
    for (int kx = 0; kx < 4; ++kx) {
        const int kk = kx << 5;
        union { v16bf_t v; u16x8_t h[2]; } a, b;
        // A frag: lane holds row r; u[0..7] = k in [kk+half*16, +8),
        //                           u[8..15] = k in [kk+half*16+8, +8)
        a.h[0] = *(const u16x8_t*)(a1s + r * APITCH + kk + half * 16);
        a.h[1] = *(const u16x8_t*)(a1s + r * APITCH + kk + half * 16 + 8);
        // B frag: pre-packed, two contiguous b128 loads
        const unsigned short* bp = W1p + ((((wave * 4 + kx) * 2 + half) * 16 + r) << 4);
        if (kx < 3) __builtin_prefetch(bp + 512, 0, 3);   // next K-slab
        b.h[0] = *(const u16x8_t*)(bp);
        b.h[1] = *(const u16x8_t*)(bp + 8);
        acc = __builtin_amdgcn_wmma_f32_16x16x32_bf16(
            false, a.v, false, b.v, (short)0, acc, false, false);
        if (HASA2) {
            a.h[0] = *(const u16x8_t*)(a2s + r * APITCH + kk + half * 16);
            a.h[1] = *(const u16x8_t*)(a2s + r * APITCH + kk + half * 16 + 8);
            const unsigned short* bp2 = W2p + ((((wave * 4 + kx) * 2 + half) * 16 + r) << 4);
            b.h[0] = *(const u16x8_t*)(bp2);
            b.h[1] = *(const u16x8_t*)(bp2 + 8);
            acc = __builtin_amdgcn_wmma_f32_16x16x32_bf16(
                false, a.v, false, b.v, (short)0, acc, false, false);
        }
    }

    // ---- epilogue: bias + optional ReLU; f32 C/D layout row = m0+half*8+i ----
    const float bv = bias[n0 + r];
#pragma unroll
    for (int i = 0; i < 8; ++i) {
        const int orow = m0 + half * 8 + i;
        if (orow < nrows) {
            float v = acc[i] + bv;
            if (RELU) v = fmaxf(v, 0.0f);
            out[(long)orow * DD + n0 + r] = v;
        }
    }
}

// ---------------------------------------------------------------------------
// 2-token multihead attention + mean over tokens (wave32 shuffle reductions).
// Writes osum = 0.5*(o0+o1); Wo/bo applied afterwards (linearity).
// ---------------------------------------------------------------------------
__global__ void attn_pool_kernel(const float* __restrict__ q0, const float* __restrict__ q1,
                                 const float* __restrict__ k0, const float* __restrict__ k1,
                                 const float* __restrict__ v0, const float* __restrict__ v1,
                                 float* __restrict__ osum, int n) {
    long gid  = (long)blockIdx.x * blockDim.x + threadIdx.x;
    long node = gid >> 5;
    int  lane = (int)(gid & 31);
    if (node >= n) return;
    const float inv_sqrt_dh = 0.17677669529663687f; // 1/sqrt(32)
#pragma unroll
    for (int h = 0; h < 4; ++h) {
        long off = node * DD + h * 32 + lane;
        float Q0 = q0[off], Q1 = q1[off];
        float K0 = k0[off], K1 = k1[off];
        float V0 = v0[off], V1 = v1[off];
        float s00 = Q0 * K0, s01 = Q0 * K1, s10 = Q1 * K0, s11 = Q1 * K1;
#pragma unroll
        for (int d = 16; d > 0; d >>= 1) {
            s00 += __shfl_xor(s00, d, 32);
            s01 += __shfl_xor(s01, d, 32);
            s10 += __shfl_xor(s10, d, 32);
            s11 += __shfl_xor(s11, d, 32);
        }
        s00 *= inv_sqrt_dh; s01 *= inv_sqrt_dh;
        s10 *= inv_sqrt_dh; s11 *= inv_sqrt_dh;
        float m0 = fmaxf(s00, s01), m1 = fmaxf(s10, s11);
        float e00 = __expf(s00 - m0), e01 = __expf(s01 - m0);
        float e10 = __expf(s10 - m1), e11 = __expf(s11 - m1);
        float o0 = (e00 * V0 + e01 * V1) / (e00 + e01);
        float o1 = (e10 * V0 + e11 * V1) / (e10 + e11);
        osum[off] = 0.5f * (o0 + o1);
    }
}

// out[node] = dot(h[node,:], W2[:,0]) + b2
__global__ void final_dot_kernel(const float* __restrict__ h,
                                 const float* __restrict__ W2,
                                 const float* __restrict__ b2,
                                 float* __restrict__ out, int n) {
    long gid  = (long)blockIdx.x * blockDim.x + threadIdx.x;
    long node = gid >> 5;
    int  lane = (int)(gid & 31);
    if (node >= n) return;
    const float* hr = h + node * DD;
    float acc = 0.0f;
#pragma unroll
    for (int k = 0; k < DD; k += 32) acc += hr[k + lane] * W2[k + lane];
#pragma unroll
    for (int d = 16; d > 0; d >>= 1) acc += __shfl_xor(acc, d, 32);
    if (lane == 0) out[node] = acc + b2[0];
}

// ---------------------------------------------------------------------------
// Host orchestration
// ---------------------------------------------------------------------------
static void launch_gemm(const float* A1, const float* cnt, const float* A2,
                        const unsigned short* W1p, const unsigned short* W2p,
                        const float* bias, float* out, int nrows, int relu,
                        hipStream_t s) {
    unsigned blocks = (unsigned)((nrows + 15) >> 4);   // one block per m-tile
    if (A2) {  // SAGE conv path: fused mean + dual matmul + ReLU
        gemm_bias_act_kernel<true, true, true>
            <<<blocks, 256, 0, s>>>(A1, cnt, A2, W1p, W2p, bias, out, nrows);
    } else if (relu) {
        gemm_bias_act_kernel<false, false, true>
            <<<blocks, 256, 0, s>>>(A1, nullptr, nullptr, W1p, nullptr, bias, out, nrows);
    } else {
        gemm_bias_act_kernel<false, false, false>
            <<<blocks, 256, 0, s>>>(A1, nullptr, nullptr, W1p, nullptr, bias, out, nrows);
    }
}

static void launch_zero(float* p, long n, hipStream_t s) {
    long blocks = (n + 255) / 256;
    zero_kernel<<<(unsigned)blocks, 256, 0, s>>>(p, n);
}

extern "C" void kernel_launch(void* const* d_in, const int* in_sizes, int n_in,
                              void* d_out, int out_size, void* d_ws, size_t ws_size,
                              hipStream_t stream) {
    const int n = in_sizes[0] / DD;          // nodes per type (100000)
    const int e = in_sizes[3] / 2;           // edges per relation (800000)

    const float* x_driver = (const float*)d_in[0];
    const float* x_race   = (const float*)d_in[1];
    const float* x_result = (const float*)d_in[2];
    const int*   ei[4]    = {(const int*)d_in[3], (const int*)d_in[4],
                             (const int*)d_in[5], (const int*)d_in[6]};
    const float* Wl[2][2] = {{(const float*)d_in[7],  (const float*)d_in[10]},
                             {(const float*)d_in[15], (const float*)d_in[18]}};
    const float* bl[2][2] = {{(const float*)d_in[8],  (const float*)d_in[11]},
                             {(const float*)d_in[16], (const float*)d_in[19]}};
    const float* Wr[2][2] = {{(const float*)d_in[9],  (const float*)d_in[12]},
                             {(const float*)d_in[17], (const float*)d_in[20]}};
    const float* outW[2]  = {(const float*)d_in[13], (const float*)d_in[21]};
    const float* outb[2]  = {(const float*)d_in[14], (const float*)d_in[22]};
    const float* Wq = (const float*)d_in[23];
    const float* Wk = (const float*)d_in[24];
    const float* Wv = (const float*)d_in[25];
    const float* Wo = (const float*)d_in[26];
    const float* W1 = (const float*)d_in[27];
    const float* bq = (const float*)d_in[28];
    const float* bk = (const float*)d_in[29];
    const float* bv = (const float*)d_in[30];
    const float* bo = (const float*)d_in[31];
    const float* b1 = (const float*)d_in[32];
    const float* W2 = (const float*)d_in[33];
    const float* b2 = (const float*)d_in[34];

    // --- workspace layout: 9 [N,D] f32 buffers + cnt + packed bf16 weights ---
    const size_t NF = (size_t)n * DD;
    char* ws = (char*)d_ws;
    float* B[9];
    for (int i = 0; i < 9; ++i) B[i] = (float*)(ws + (size_t)i * NF * 4);
    float* cnt = (float*)(ws + 9 * NF * 4);
    unsigned short* wbf = (unsigned short*)(ws + 9 * NF * 4 + (size_t)n * 4);
    const int WM = DD * DD;

    // packed slots: 0..3 Wl/Wr m0c0,m0c1 ; 4..7 m1 ; 8,9 outW ; 10..14 Wq,Wk,Wv,Wo,W1
    const float* wsrc[15] = {Wl[0][0], Wr[0][0], Wl[0][1], Wr[0][1],
                             Wl[1][0], Wr[1][0], Wl[1][1], Wr[1][1],
                             outW[0], outW[1], Wq, Wk, Wv, Wo, W1};
    for (int i = 0; i < 15; ++i)
        pack_w_kernel<<<WM / 256, 256, 0, stream>>>(wsrc[i], wbf + (size_t)i * WM);

    const unsigned sc_blocks = (unsigned)(((long)e * 32 + 255) / 256);

    float* agg = B[0];
    const float* xm[2] = {x_race, x_result};
    float* emb[2]      = {B[3], B[4]};

    // --- two metapaths: scatter-mean + fused SAGE GEMMs + out projection ---
    for (int m = 0; m < 2; ++m) {
        const unsigned short* wl0 = wbf + (size_t)(4 * m + 0) * WM;
        const unsigned short* wr0 = wbf + (size_t)(4 * m + 1) * WM;
        const unsigned short* wl1 = wbf + (size_t)(4 * m + 2) * WM;
        const unsigned short* wr1 = wbf + (size_t)(4 * m + 3) * WM;
        const unsigned short* wo_ = wbf + (size_t)(8 + m) * WM;

        // hop 1: driver -> mid
        launch_zero(agg, (long)NF, stream);
        launch_zero(cnt, n, stream);
        scatter_accum_kernel<<<sc_blocks, 256, 0, stream>>>(
            x_driver, ei[2 * m], ei[2 * m] + e, agg, cnt, e);
        launch_gemm(agg, cnt, xm[m], wl0, wr0, bl[m][0], B[1], n, 1, stream);    // h_mid

        // hop 2: mid -> driver
        launch_zero(agg, (long)NF, stream);
        launch_zero(cnt, n, stream);
        scatter_accum_kernel<<<sc_blocks, 256, 0, stream>>>(
            B[1], ei[2 * m + 1], ei[2 * m + 1] + e, agg, cnt, e);
        launch_gemm(agg, cnt, x_driver, wl1, wr1, bl[m][1], B[2], n, 1, stream); // h_start

        launch_gemm(B[2], nullptr, nullptr, wo_, nullptr, outb[m], emb[m], n, 0, stream);
    }

    // --- attention head: QKV projections (WMMA), 2-token attention, MLP ---
    const unsigned short* wqb = wbf + (size_t)10 * WM;
    const unsigned short* wkb = wbf + (size_t)11 * WM;
    const unsigned short* wvb = wbf + (size_t)12 * WM;
    const unsigned short* wob = wbf + (size_t)13 * WM;
    const unsigned short* w1b = wbf + (size_t)14 * WM;

    launch_gemm(emb[0], nullptr, nullptr, wqb, nullptr, bq, B[0], n, 0, stream); // q0
    launch_gemm(emb[1], nullptr, nullptr, wqb, nullptr, bq, B[1], n, 0, stream); // q1
    launch_gemm(emb[0], nullptr, nullptr, wkb, nullptr, bk, B[2], n, 0, stream); // k0
    launch_gemm(emb[1], nullptr, nullptr, wkb, nullptr, bk, B[5], n, 0, stream); // k1
    launch_gemm(emb[0], nullptr, nullptr, wvb, nullptr, bv, B[6], n, 0, stream); // v0
    launch_gemm(emb[1], nullptr, nullptr, wvb, nullptr, bv, B[7], n, 0, stream); // v1

    const unsigned at_blocks = (unsigned)(((long)n * 32 + 255) / 256);
    attn_pool_kernel<<<at_blocks, 256, 0, stream>>>(B[0], B[1], B[2], B[5],
                                                    B[6], B[7], B[8], n);

    launch_gemm(B[8], nullptr, nullptr, wob, nullptr, bo, B[0], n, 0, stream);   // pooled
    launch_gemm(B[0], nullptr, nullptr, w1b, nullptr, b1, B[1], n, 1, stream);   // hidden

    final_dot_kernel<<<at_blocks, 256, 0, stream>>>(B[1], W2, b2, (float*)d_out, n);
}
